// BPStyleBlock_69011534512920
// MI455X (gfx1250) — matrix operand order repaired and checked by reading, not compiled, not run on hardware
//
#include <hip/hip_runtime.h>
#include <hip/hip_bf16.h>

// ---------------------------------------------------------------------------
// BPStyleBlock for MI455X (gfx1250): all matmul-shaped work on
// v_wmma_f32_16x16x32_f16 (f16 in, f32 accumulate), wave32 fragment layouts
// with A/B K contiguous per lane half (k = 16h + s).
// ---------------------------------------------------------------------------

typedef _Float16 f16;
typedef f16  f16x8  __attribute__((ext_vector_type(8)));
typedef f16  f16x16 __attribute__((ext_vector_type(16)));
typedef float f32x8 __attribute__((ext_vector_type(8)));

#define NN   16
#define CC   128
#define TT   256
#define TSS  128
#define VV   24
#define LTV  (TT*VV)          // 6144
#define LSV  (TSS*VV)         // 3072
#define PADR ((TT+2)*VV)      // 6192 padded rows for tcn
#define SPERN 3211264         // 32768 * 98 = per-n attention matrix elems

// part tables: ORDER = spine, leftleg, leftarm, neck, rightarm, rightleg
__constant__ int c_nodes_[24] = {0,5,6,7,8, 1,2,3,4, 9,10,11,12, 13,14,15,
                                 16,17,18,19, 20,21,22,23};
__constant__ int c_cum_[7]   = {0,5,9,13,16,20,24};
__constant__ int c_cum2_[7]  = {0,25,41,57,66,82,98};      // cumsum vp^2
__constant__ int c_partof_[24] = {0,1,1,1,1,0,0,0,0,2,2,2,2,3,3,3,
                                  4,4,4,4,5,5,5,5};

// ---------------- WMMA fragment helpers (contiguous-K A/B layouts) ---------------
__device__ __forceinline__ f16x16 mk16(f16x8 lo, f16x8 hi) {
  f16x16 r;
#pragma unroll
  for (int i = 0; i < 8; ++i) { r[i] = lo[i]; r[8 + i] = hi[i]; }
  return r;
}
// A (16x32 MxK): lane m=lane&15, h=lane>>4; K contiguous [16h .. 16h+15]
__device__ __forceinline__ f16x16 load_afrag(const f16* p) {
  return mk16(*(const f16x8*)p, *(const f16x8*)(p + 8));
}
// B (32x16 KxN): lane n=lane&15, h=lane>>4; K contiguous [16h .. 16h+15]
__device__ __forceinline__ f16x16 load_bfrag(const f16* p) {
  return mk16(*(const f16x8*)p, *(const f16x8*)(p + 8));
}
#define WMMA(a,b,c) __builtin_amdgcn_wmma_f32_16x16x32_f16(false,(a),false,(b),(short)0,(c),false,false)

// ---------------------------- small utility kernels ------------------------
__global__ void k_cvt(f16* __restrict__ dst, const float* __restrict__ src, int cnt) {
  int i = blockIdx.x * 256 + threadIdx.x;
  if (i < cnt) dst[i] = (f16)src[i];
}

// tcn weight (O,C,3,1) -> f16 [o][dt*128+c]
__global__ void k_reorder_tcn(f16* __restrict__ dst, const float* __restrict__ src) {
  int i = blockIdx.x * 256 + threadIdx.x;
  if (i >= 128 * 384) return;
  int o = i / 384, k = i % 384, dt = k >> 7, c = k & 127;
  dst[i] = (f16)src[(o * 128 + c) * 3 + dt];
}

// per (part, n, c): mean / rsqrt(var+eps) over (Tdim, part nodes)
__global__ void k_stats(const float* __restrict__ src, int Tdim,
                        float* __restrict__ out, int pfix) {
  int c = blockIdx.x, n = blockIdx.y;
  int p = (pfix >= 0) ? pfix : (int)blockIdx.z;
  int b0 = c_cum_[p], vp = c_cum_[p + 1] - b0;
  int cnt = Tdim * vp;
  const float* b = src + (long)(n * CC + c) * Tdim * VV;
  float s = 0.f, s2 = 0.f;
  for (int e = threadIdx.x; e < cnt; e += blockDim.x) {
    int t = e / vp, vl = e - t * vp;
    float x = b[t * VV + c_nodes_[b0 + vl]];
    s += x; s2 += x * x;
  }
  for (int o = 16; o > 0; o >>= 1) { s += __shfl_down(s, o, 32); s2 += __shfl_down(s2, o, 32); }
  __shared__ float r0[4], r1[4];
  int w = threadIdx.x >> 5;
  if ((threadIdx.x & 31) == 0) { r0[w] = s; r1[w] = s2; }
  __syncthreads();
  if (threadIdx.x == 0) {
    float S = r0[0]+r0[1]+r0[2]+r0[3], S2 = r1[0]+r1[1]+r1[2]+r1[3];
    float m = S / cnt, var = S2 / cnt - m * m;
    float* op = out + (((long)p * NN + n) * CC + c) * 2;
    op[0] = m; op[1] = rsqrtf(var + 1e-5f);
  }
}

// adain style MLP: smean(128) -> 64 -> 64 -> 256 (g|b); one block per n
__global__ void k_adain_mlp(const float* __restrict__ sstats, int p,
                            const float* __restrict__ tlw, const float* __restrict__ tlb,
                            const float* __restrict__ i1w, const float* __restrict__ i1b,
                            const float* __restrict__ i2w, const float* __restrict__ i2b,
                            float* __restrict__ gb) {
  int n = blockIdx.x, j = threadIdx.x;           // 64 threads
  __shared__ float sm[128], z[64], hh[64];
  const float* st = sstats + ((long)p * NN + n) * CC * 2;
  for (int c = j; c < 128; c += 64) sm[c] = st[c * 2];
  __syncthreads();
  float a = tlb[j];
  for (int c = 0; c < 128; ++c) a += sm[c] * tlw[j * 128 + c];
  z[j] = a >= 0.f ? a : 0.2f * a;
  __syncthreads();
  float b = i1b[j];
  for (int k = 0; k < 64; ++k) b += z[k] * i1w[j * 64 + k];
  hh[j] = b >= 0.f ? b : 0.2f * b;
  __syncthreads();
  for (int r = 0; r < 4; ++r) {
    int o = r * 64 + j;
    float v = i2b[o];
    for (int k = 0; k < 64; ++k) v += hh[k] * i2w[o * 64 + k];
    gb[((long)p * NN + n) * 256 + o] = v;
  }
}

// adain + lrelu; writes x1 transposed f16: x1h[n][t*V+v][c]
__global__ void k_adain_apply(const float* __restrict__ x, const float* __restrict__ xstats,
                              const float* __restrict__ gb, f16* __restrict__ x1h) {
  int l = blockIdx.x, n = blockIdx.y, c = threadIdx.x;
  int v = l % VV, t = l / VV;
  int p = c_partof_[v];
  float xv = x[((long)(n * CC + c) * TT + t) * VV + v];
  const float* st = xstats + (((long)p * NN + n) * CC + c) * 2;
  float g = gb[((long)p * NN + n) * 256 + c];
  float b = gb[((long)p * NN + n) * 256 + 128 + c];
  float y = (1.f + g) * (xv - st[0]) * st[1] + b;
  y = y >= 0.f ? y : 0.2f * y;
  x1h[((long)n * LTV + l) * 128 + c] = (f16)y;
}

// adjacency einsum y(n,3*128,t,v) x A(3,24,24) -> writes transposed, padded
// f16 buffer for the tcn (reflect pad rows 0 and T+1 fused in)
__global__ void k_einsum(const float* __restrict__ y, const float* __restrict__ A,
                         f16* __restrict__ xpad) {
  __shared__ float As[1728];
  for (int i = threadIdx.x; i < 1728; i += blockDim.x) As[i] = A[i];
  __syncthreads();
  long gid = (long)blockIdx.x * blockDim.x + threadIdx.x;
  int w = gid % VV; long r = gid / VV;
  int t = r % TT; r /= TT;
  int c = r % CC; int n = (int)(r / CC);
  float acc = 0.f;
#pragma unroll
  for (int k = 0; k < 3; ++k) {
    const float* yp = y + ((long)(n * 384 + k * 128 + c) * TT + t) * VV;
    const float* Ap = As + k * 576 + w;
#pragma unroll
    for (int v = 0; v < VV; ++v) acc += yp[v] * Ap[v * VV];
  }
  f16 hv = (f16)acc;
  long pb = (long)n * PADR * 128;
  xpad[pb + ((long)(t + 1) * VV + w) * 128 + c] = hv;
  if (t == 1)      xpad[pb + ((long)0 * VV + w) * 128 + c] = hv;      // reflect left
  if (t == TT - 2) xpad[pb + ((long)(TT + 1) * VV + w) * 128 + c] = hv; // reflect right
}

// pack inorm(x3) per part, transposed f16 (rows = packed (part,t,vl), cols = c)
__global__ void k_gather_x(const float* __restrict__ x3, const float* __restrict__ xstats,
                           f16* __restrict__ xg) {
  int row = blockIdx.x, n = blockIdx.y, c = threadIdx.x;
  int p = 0; while (row >= c_cum_[p + 1] * TT) ++p;
  int vp = c_cum_[p + 1] - c_cum_[p];
  int rl = row - c_cum_[p] * TT;
  int t = rl / vp, vl = rl - t * vp;
  int vg = c_nodes_[c_cum_[p] + vl];
  float xv = x3[((long)(n * CC + c) * TT + t) * VV + vg];
  const float* st = xstats + (((long)p * NN + n) * CC + c) * 2;
  xg[((long)n * LTV + row) * 128 + c] = (f16)((xv - st[0]) * st[1]);
}

// pack style: raw (for H conv) and inorm'd (for G conv), transposed f16
__global__ void k_gather_s(const float* s0, const float* s1, const float* s2,
                           const float* s3, const float* s4, const float* s5,
                           const float* __restrict__ sstats,
                           f16* __restrict__ sgn, f16* __restrict__ sgr) {
  int row = blockIdx.x, n = blockIdx.y, c = threadIdx.x;
  int p = 0; while (row >= c_cum_[p + 1] * TSS) ++p;
  int vp = c_cum_[p + 1] - c_cum_[p];
  int rl = row - c_cum_[p] * TSS;
  int t = rl / vp, vl = rl - t * vp;
  int vg = c_nodes_[c_cum_[p] + vl];
  const float* sp = (p == 0) ? s0 : (p == 1) ? s1 : (p == 2) ? s2
                   : (p == 3) ? s3 : (p == 4) ? s4 : s5;
  float sv = sp[((long)(n * CC + c) * TSS + t) * VV + vg];
  const float* st = sstats + (((long)p * NN + n) * CC + c) * 2;
  long o = ((long)n * LSV + row) * 128 + c;
  sgr[o] = (f16)sv;
  sgn[o] = (f16)((sv - st[0]) * st[1]);
}

// row softmax over L2, f32 in -> f16 out (same layout)
__global__ void k_softmax(const float* __restrict__ S32, f16* __restrict__ S16) {
  int row = blockIdx.x;
  int n = row / LTV, r = row % LTV;
  int p = 0; while (r >= c_cum_[p + 1] * TT) ++p;
  int vp = c_cum_[p + 1] - c_cum_[p];
  int l1 = r - c_cum_[p] * TT;
  int L2 = TSS * vp;
  long base = (long)n * SPERN + (long)32768 * c_cum2_[p] + (long)l1 * L2;
  const float* src = S32 + base;
  float mx = -1e30f;
  for (int i = threadIdx.x; i < L2; i += blockDim.x) mx = fmaxf(mx, src[i]);
  for (int o = 16; o > 0; o >>= 1) mx = fmaxf(mx, __shfl_xor(mx, o, 32));
  __shared__ float rm[4], rs[4];
  int w = threadIdx.x >> 5;
  if ((threadIdx.x & 31) == 0) rm[w] = mx;
  __syncthreads();
  mx = fmaxf(fmaxf(rm[0], rm[1]), fmaxf(rm[2], rm[3]));
  float sm = 0.f;
  for (int i = threadIdx.x; i < L2; i += blockDim.x) sm += __expf(src[i] - mx);
  for (int o = 16; o > 0; o >>= 1) sm += __shfl_xor(sm, o, 32);
  if ((threadIdx.x & 31) == 0) rs[w] = sm;
  __syncthreads();
  sm = rs[0] + rs[1] + rs[2] + rs[3];
  float inv = 1.f / sm;
  f16* dst = S16 + base;
  for (int i = threadIdx.x; i < L2; i += blockDim.x)
    dst[i] = (f16)(__expf(src[i] - mx) * inv);
}

// --------------------------- WMMA GEMM kernels -----------------------------
// All: block = 128 thr = 4 waves; wave tile 16(M) x 64(N); K step 32.

// D(f32, M-major) = W(MxK) @ Bt(LxK)^T + bias ; batched over z
__global__ void k_gemm_plain(const f16* __restrict__ A, const f16* __restrict__ Bt,
                             const float* __restrict__ bias, float* __restrict__ D,
                             int K, int ldD, long strideB, long strideD) {
  int lane = threadIdx.x & 31, wv = threadIdx.x >> 5;
  int m0 = blockIdx.y * 64 + wv * 16, n0 = blockIdx.x * 64, z = blockIdx.z;
  Bt += (long)z * strideB; D += (long)z * strideD;
  int mr = lane & 15, h = lane >> 4;
  const f16* ap = A + (long)(m0 + mr) * K + 16 * h;
  const f16* bp = Bt + (long)(n0 + mr) * K + 16 * h;
  f32x8 acc[4] = {};
  for (int k0 = 0; k0 < K; k0 += 32) {
    f16x16 af = load_afrag(ap + k0);
#pragma unroll
    for (int j = 0; j < 4; ++j) {
      f16x16 bf = load_bfrag(bp + (long)j * 16 * K + k0);
      acc[j] = WMMA(af, bf, acc[j]);
    }
  }
#pragma unroll
  for (int j = 0; j < 4; ++j)
#pragma unroll
    for (int r = 0; r < 8; ++r) {
      int m = m0 + r + 8 * h;
      D[(long)m * ldD + n0 + j * 16 + mr] = acc[j][r] + bias[m];
    }
}

// temporal conv as K=384 GEMM over padded/transposed input (dt row offsets)
__global__ void k_gemm_tcn(const f16* __restrict__ A, const f16* __restrict__ Bt,
                           const float* __restrict__ bias, float* __restrict__ D) {
  int lane = threadIdx.x & 31, wv = threadIdx.x >> 5;
  int m0 = blockIdx.y * 64 + wv * 16, n0 = blockIdx.x * 64, z = blockIdx.z;
  const f16* B = Bt + (long)z * PADR * 128;
  float* Dz = D + (long)z * CC * LTV;
  int mr = lane & 15, h = lane >> 4;
  const f16* ap = A + (long)(m0 + mr) * 384 + 16 * h;
  f32x8 acc[4] = {};
  for (int k0 = 0; k0 < 384; k0 += 32) {
    int dt = k0 >> 7, kc = k0 & 127;
    f16x16 af = load_afrag(ap + k0);
    const f16* bp = B + (long)(n0 + mr + dt * VV) * 128 + kc + 16 * h;
#pragma unroll
    for (int j = 0; j < 4; ++j) {
      f16x16 bf = load_bfrag(bp + (long)j * 16 * 128);
      acc[j] = WMMA(af, bf, acc[j]);
    }
  }
#pragma unroll
  for (int j = 0; j < 4; ++j)
#pragma unroll
    for (int r = 0; r < 8; ++r) {
      int m = m0 + r + 8 * h;
      Dz[(long)m * LTV + n0 + j * 16 + mr] = acc[j][r] + bias[m];
    }
}

// per-(n,part) 1x1 conv, output transposed f16 (L x 128) : F and G paths
__global__ void k_gemm_conv(const f16* __restrict__ Wq, const float* __restrict__ biasq,
                            const f16* __restrict__ src, f16* __restrict__ dst, int tdim) {
  int z = blockIdx.z, p = z % 6, n = z / 6;
  int vp = c_cum_[p + 1] - c_cum_[p];
  int Lp = tdim * vp;
  int n0 = blockIdx.x * 64;
  if (n0 >= Lp) return;
  int lane = threadIdx.x & 31, wv = threadIdx.x >> 5;
  int m0 = blockIdx.y * 64 + wv * 16;
  long rowbase = (long)n * VV * tdim + (long)c_cum_[p] * tdim + n0;
  const f16* Bt = src + rowbase * 128;
  f16* Dt = dst + rowbase * 128;
  const f16* A = Wq + (long)p * 65536;
  const float* bias = biasq + p * 512;
  int mr = lane & 15, h = lane >> 4;
  const f16* ap = A + (long)(m0 + mr) * 128 + 16 * h;
  const f16* bp = Bt + (long)mr * 128 + 16 * h;
  f32x8 acc[4] = {};
  for (int k0 = 0; k0 < 128; k0 += 32) {
    f16x16 af = load_afrag(ap + k0);
#pragma unroll
    for (int j = 0; j < 4; ++j) {
      f16x16 bf = load_bfrag(bp + (long)j * 16 * 128 + k0);
      acc[j] = WMMA(af, bf, acc[j]);
    }
  }
#pragma unroll
  for (int j = 0; j < 4; ++j)
#pragma unroll
    for (int r = 0; r < 8; ++r) {
      int m = m0 + r + 8 * h;
      Dt[(long)(j * 16 + mr) * 128 + m] = (f16)(acc[j][r] + bias[m]);
    }
}

// H conv: output f16 M-major with row stride 3072 (per-n 128 x 3072 packed)
__global__ void k_gemm_convH(const f16* __restrict__ Wq, const float* __restrict__ biasq,
                             const f16* __restrict__ src, f16* __restrict__ Hh) {
  int z = blockIdx.z, p = z % 6, n = z / 6;
  int vp = c_cum_[p + 1] - c_cum_[p];
  int Lp = TSS * vp;
  int n0 = blockIdx.x * 64;
  if (n0 >= Lp) return;
  int lane = threadIdx.x & 31, wv = threadIdx.x >> 5;
  int m0 = blockIdx.y * 64 + wv * 16;
  const f16* Bt = src + ((long)n * LSV + (long)c_cum_[p] * TSS + n0) * 128;
  f16* D = Hh + (long)n * 128 * LSV + (long)c_cum_[p] * TSS + n0;
  const f16* A = Wq + (long)p * 65536;
  const float* bias = biasq + p * 512;
  int mr = lane & 15, h = lane >> 4;
  const f16* ap = A + (long)(m0 + mr) * 128 + 16 * h;
  const f16* bp = Bt + (long)mr * 128 + 16 * h;
  f32x8 acc[4] = {};
  for (int k0 = 0; k0 < 128; k0 += 32) {
    f16x16 af = load_afrag(ap + k0);
#pragma unroll
    for (int j = 0; j < 4; ++j) {
      f16x16 bf = load_bfrag(bp + (long)j * 16 * 128 + k0);
      acc[j] = WMMA(af, bf, acc[j]);
    }
  }
#pragma unroll
  for (int j = 0; j < 4; ++j)
#pragma unroll
    for (int r = 0; r < 8; ++r) {
      int m = m0 + r + 8 * h;
      D[(long)m * LSV + j * 16 + mr] = (f16)(acc[j][r] + bias[m]);
    }
}

// S = F^T G : (L1 x L2) f32, K = 128
__global__ void k_gemm_S(const f16* __restrict__ Fh, const f16* __restrict__ Gh,
                         float* __restrict__ S32) {
  int z = blockIdx.z, p = z % 6, n = z / 6;
  int vp = c_cum_[p + 1] - c_cum_[p];
  int L1 = TT * vp, L2 = TSS * vp;
  int lane = threadIdx.x & 31, wv = threadIdx.x >> 5;
  int m0 = blockIdx.y * 64 + wv * 16;
  int n0 = blockIdx.x * 64;
  if (m0 >= L1 || n0 >= L2) return;
  const f16* A = Fh + ((long)n * LTV + (long)c_cum_[p] * TT) * 128;
  const f16* Bt = Gh + ((long)n * LSV + (long)c_cum_[p] * TSS) * 128;
  float* D = S32 + (long)n * SPERN + (long)32768 * c_cum2_[p];
  int mr = lane & 15, h = lane >> 4;
  const f16* ap = A + (long)(m0 + mr) * 128 + 16 * h;
  const f16* bp = Bt + (long)(n0 + mr) * 128 + 16 * h;
  f32x8 acc[4] = {};
  for (int k0 = 0; k0 < 128; k0 += 32) {
    f16x16 af = load_afrag(ap + k0);
#pragma unroll
    for (int j = 0; j < 4; ++j) {
      f16x16 bf = load_bfrag(bp + (long)j * 16 * 128 + k0);
      acc[j] = WMMA(af, bf, acc[j]);
    }
  }
#pragma unroll
  for (int j = 0; j < 4; ++j)
#pragma unroll
    for (int r = 0; r < 8; ++r)
      D[(long)(m0 + r + 8 * h) * L2 + n0 + j * 16 + mr] = acc[j][r];
}

// O = H @ S^T : M=128, N=L1, K=L2; output transposed f16 (L1 x 128)
__global__ void k_gemm_O(const f16* __restrict__ Hh, const f16* __restrict__ S16,
                         f16* __restrict__ Oh) {
  int z = blockIdx.z, p = z % 6, n = z / 6;
  int vp = c_cum_[p + 1] - c_cum_[p];
  int L1 = TT * vp, L2 = TSS * vp;
  int n0 = blockIdx.x * 64;
  if (n0 >= L1) return;
  int lane = threadIdx.x & 31, wv = threadIdx.x >> 5;
  int m0 = blockIdx.y * 64 + wv * 16;
  const f16* A = Hh + (long)n * 128 * LSV + (long)c_cum_[p] * TSS;       // lda = LSV
  const f16* Bt = S16 + (long)n * SPERN + (long)32768 * c_cum2_[p];     // ldb = L2
  f16* Dt = Oh + ((long)n * LTV + (long)c_cum_[p] * TT + n0) * 128;
  int mr = lane & 15, h = lane >> 4;
  const f16* ap = A + (long)(m0 + mr) * LSV + 16 * h;
  const f16* bp = Bt + (long)(n0 + mr) * L2 + 16 * h;
  f32x8 acc[4] = {};
  for (int k0 = 0; k0 < L2; k0 += 32) {
    f16x16 af = load_afrag(ap + k0);
#pragma unroll
    for (int j = 0; j < 4; ++j) {
      f16x16 bf = load_bfrag(bp + (long)j * 16 * L2 + k0);
      acc[j] = WMMA(af, bf, acc[j]);
    }
  }
#pragma unroll
  for (int j = 0; j < 4; ++j)
#pragma unroll
    for (int r = 0; r < 8; ++r) {
      int m = m0 + r + 8 * h;
      Dt[(long)(j * 16 + mr) * 128 + m] = (f16)acc[j][r];
    }
}

// out-proj conv + bias + residual; scatters f16 into (l=t*V+vg, c) layout
__global__ void k_gemm_Kout(const f16* __restrict__ Wq, const float* __restrict__ biasq,
                            const f16* __restrict__ Oh, const float* __restrict__ x3,
                            f16* __restrict__ x4h) {
  int z = blockIdx.z, p = z % 6, n = z / 6;
  int vp = c_cum_[p + 1] - c_cum_[p];
  int L1 = TT * vp;
  int n0 = blockIdx.x * 64;
  if (n0 >= L1) return;
  int lane = threadIdx.x & 31, wv = threadIdx.x >> 5;
  int m0 = blockIdx.y * 64 + wv * 16;
  const f16* A = Wq + (long)p * 65536;
  const float* bias = biasq + p * 512;
  const f16* Bt = Oh + ((long)n * LTV + (long)c_cum_[p] * TT + n0) * 128;
  int mr = lane & 15, h = lane >> 4;
  const f16* ap = A + (long)(m0 + mr) * 128 + 16 * h;
  const f16* bp = Bt + (long)mr * 128 + 16 * h;
  f32x8 acc[4] = {};
  for (int k0 = 0; k0 < 128; k0 += 32) {
    f16x16 af = load_afrag(ap + k0);
#pragma unroll
    for (int j = 0; j < 4; ++j) {
      f16x16 bf = load_bfrag(bp + (long)j * 16 * 128 + k0);
      acc[j] = WMMA(af, bf, acc[j]);
    }
  }
#pragma unroll
  for (int j = 0; j < 4; ++j)
#pragma unroll
    for (int r = 0; r < 8; ++r) {
      int m = m0 + r + 8 * h;
      int l1 = n0 + j * 16 + mr;
      int t = l1 / vp, vl = l1 - t * vp;
      int vg = c_nodes_[c_cum_[p] + vl];
      float res = x3[((long)(n * CC + m) * TT + t) * VV + vg];
      x4h[((long)n * LTV + (long)t * VV + vg) * 128 + m] = (f16)(acc[j][r] + bias[m] + res);
    }
}

// ------------------------------- host side --------------------------------
extern "C" void kernel_launch(void* const* d_in, const int* in_sizes, int n_in,
                              void* d_out, int out_size, void* d_ws, size_t ws_size,
                              hipStream_t stream) {
  (void)in_sizes; (void)n_in; (void)out_size; (void)ws_size;
  // inputs in setup_inputs() dict-insertion order:
  // 0:x 1..6:s_* 7:A  8..43: adain (ORDER x {tl_w,tl_b,i1_w,i1_b,i2_w,i2_b})
  // 44..91: cross (ORDER x {f_w,f_b,g_w,g_b,h_w,h_b,k_w,k_b})
  // 92:gcn1_w 93:gcn1_b 94:tcn1_w 95:tcn1_b 96:gcn2_w 97:gcn2_b 98:tcn2_w 99:tcn2_b
  const float* X = (const float*)d_in[0];
  const float* Sty[6];
  for (int p = 0; p < 6; ++p) Sty[p] = (const float*)d_in[1 + p];
  const float* Amat = (const float*)d_in[7];
  auto IN = [&](int i) { return (const float*)d_in[i]; };

  char* cur = (char*)d_ws;
  auto carve = [&](size_t bytes) -> char* {
    char* p = cur; cur += (bytes + 255) & ~(size_t)255; return p;
  };
  float* sstats  = (float*)carve((size_t)6 * NN * CC * 2 * 4);
  float* xstats1 = (float*)carve((size_t)6 * NN * CC * 2 * 4);
  float* xstats2 = (float*)carve((size_t)6 * NN * CC * 2 * 4);
  float* gb      = (float*)carve((size_t)6 * NN * 256 * 4);
  f16* wg1h = (f16*)carve(384 * 128 * 2);
  f16* wg2h = (f16*)carve(384 * 128 * 2);
  f16* wt1h = (f16*)carve(128 * 384 * 2);
  f16* wt2h = (f16*)carve(128 * 384 * 2);
  f16* wch  = (f16*)carve((size_t)6 * 4 * 16384 * 2);
  float* cb = (float*)carve((size_t)6 * 4 * 128 * 4);
  f16*   x1h  = (f16*)carve((size_t)NN * LTV * 128 * 2);     // 25 MB
  float* y1   = (float*)carve((size_t)NN * 384 * LTV * 4);   // 151 MB
  f16*   xpad = (f16*)carve((size_t)NN * PADR * 128 * 2);    // 25 MB
  float* x3   = (float*)carve((size_t)NN * CC * LTV * 4);    // 50 MB
  f16*   xg   = (f16*)carve((size_t)NN * LTV * 128 * 2);     // 25 MB
  f16*   sgn  = (f16*)carve((size_t)NN * LSV * 128 * 2);     // 12.6 MB
  f16*   sgr  = (f16*)carve((size_t)NN * LSV * 128 * 2);
  f16*   Gh   = (f16*)carve((size_t)NN * LSV * 128 * 2);
  f16*   Hh   = (f16*)carve((size_t)NN * 128 * LSV * 2);
  float* S32  = (float*)carve((size_t)NN * SPERN * 4);       // 205 MB
  f16*   x4h  = (f16*)carve((size_t)NN * LTV * 128 * 2);
  f16* Fh  = x1h;        // alias: x1h dead after gcn1
  f16* S16 = (f16*)y1;   // alias: y1 free between einsum1 and gcn2
  f16* Oh  = xg;         // alias: xg dead after F conv

  // ---- weight prep
  k_cvt<<<dim3(192), 256, 0, stream>>>(wg1h, IN(92), 49152);
  k_cvt<<<dim3(192), 256, 0, stream>>>(wg2h, IN(96), 49152);
  k_reorder_tcn<<<dim3(192), 256, 0, stream>>>(wt1h, IN(94));
  k_reorder_tcn<<<dim3(192), 256, 0, stream>>>(wt2h, IN(98));
  for (int p = 0; p < 6; ++p)
    for (int q = 0; q < 4; ++q) {
      k_cvt<<<dim3(64), 256, 0, stream>>>(wch + (p * 4 + q) * 16384, IN(44 + 8 * p + 2 * q), 16384);
      hipMemcpyAsync(cb + (p * 4 + q) * 128, (const void*)IN(44 + 8 * p + 2 * q + 1),
                     128 * sizeof(float), hipMemcpyDeviceToDevice, stream);
    }
  // ---- stats + adain
  k_stats<<<dim3(128, 16, 6), 128, 0, stream>>>(X, TT, xstats1, -1);
  for (int p = 0; p < 6; ++p)
    k_stats<<<dim3(128, 16, 1), 128, 0, stream>>>(Sty[p], TSS, sstats, p);
  for (int p = 0; p < 6; ++p)
    k_adain_mlp<<<dim3(16), 64, 0, stream>>>(sstats, p, IN(8 + 6 * p + 0), IN(8 + 6 * p + 1),
                                             IN(8 + 6 * p + 2), IN(8 + 6 * p + 3),
                                             IN(8 + 6 * p + 4), IN(8 + 6 * p + 5), gb);
  k_adain_apply<<<dim3(LTV, 16), 128, 0, stream>>>(X, xstats1, gb, x1h);
  // ---- gcn1 -> einsum(A) -> tcn1
  k_gemm_plain<<<dim3(96, 6, 16), 128, 0, stream>>>(wg1h, x1h, IN(93), y1, 128, LTV,
                                                    (long)LTV * 128, (long)384 * LTV);
  k_einsum<<<dim3(49152), 256, 0, stream>>>(y1, Amat, xpad);
  k_gemm_tcn<<<dim3(96, 2, 16), 128, 0, stream>>>(wt1h, xpad, IN(95), x3);
  // ---- cross attention
  k_stats<<<dim3(128, 16, 6), 128, 0, stream>>>(x3, TT, xstats2, -1);
  k_gather_x<<<dim3(LTV, 16), 128, 0, stream>>>(x3, xstats2, xg);
  k_gather_s<<<dim3(LSV, 16), 128, 0, stream>>>(Sty[0], Sty[1], Sty[2], Sty[3], Sty[4], Sty[5],
                                                sstats, sgn, sgr);
  k_gemm_conv<<<dim3(20, 2, 96), 128, 0, stream>>>(wch + 0 * 16384, cb + 0 * 128, xg, Fh, TT);
  k_gemm_conv<<<dim3(10, 2, 96), 128, 0, stream>>>(wch + 1 * 16384, cb + 1 * 128, sgn, Gh, TSS);
  k_gemm_convH<<<dim3(10, 2, 96), 128, 0, stream>>>(wch + 2 * 16384, cb + 2 * 128, sgr, Hh);
  k_gemm_S<<<dim3(10, 20, 96), 128, 0, stream>>>(Fh, Gh, S32);
  k_softmax<<<dim3(16 * LTV), 128, 0, stream>>>(S32, S16);
  k_gemm_O<<<dim3(20, 2, 96), 128, 0, stream>>>(Hh, S16, Oh);
  k_gemm_Kout<<<dim3(20, 2, 96), 128, 0, stream>>>(wch + 3 * 16384, cb + 3 * 128, Oh, x3, x4h);
  // ---- gcn2 -> einsum(A) -> tcn2 -> out
  k_gemm_plain<<<dim3(96, 6, 16), 128, 0, stream>>>(wg2h, x4h, IN(97), y1, 128, LTV,
                                                    (long)LTV * 128, (long)384 * LTV);
  k_einsum<<<dim3(49152), 256, 0, stream>>>(y1, Amat, xpad);
  k_gemm_tcn<<<dim3(96, 2, 16), 128, 0, stream>>>(wt2h, xpad, IN(99), (float*)d_out);
}
